// _PointnetSAModuleBase_6193342841557
// MI455X (gfx1250) — matrix operand order repaired and checked by reading, not compiled, not run on hardware
//
#include <hip/hip_runtime.h>

// ---- problem constants (match reference setup_inputs) ----
#define B_   4
#define N_   16384
#define C_   64
#define M_   1024      // NPOINT
#define NS   32        // NSAMPLE

typedef __attribute__((ext_vector_type(16))) _Float16 v16h;
typedef __attribute__((ext_vector_type(8)))  _Float16 v8h;
typedef __attribute__((ext_vector_type(8)))  float    v8f;

// ---- CDNA5 async global->LDS copy (ASYNCcnt-tracked, no VGPR round trip) ----
__device__ __forceinline__ void async_copy_dword(unsigned lds_byte_addr,
                                                 const float* gptr) {
  asm volatile("global_load_async_to_lds_b32 %0, %1, off"
               :: "v"(lds_byte_addr),
                  "v"((unsigned long long)(uintptr_t)gptr)
               : "memory");
}
__device__ __forceinline__ void wait_async0() {
  asm volatile("s_wait_asynccnt 0" ::: "memory");
}

// LDS byte address = low 32 bits of the generic pointer (aperture rule).
__device__ __forceinline__ unsigned lds_addr_of(const void* p) {
  return (unsigned)(uintptr_t)p;
}

// =====================================================================
// Kernel 1: Furthest Point Sampling.
// One workgroup per batch; entire point cloud lives in LDS (256 KB,
// async-DMA'd in), per-thread min-dist in registers, argmax via wave
// shuffles + LDS.
// =====================================================================
__global__ __launch_bounds__(1024) void fps_kernel(const float* __restrict__ xyz,
                                                   float* __restrict__ qxyz) {
  extern __shared__ __align__(16) char smem[];
  float4* pts  = (float4*)smem;                        // [16384] = 256 KB
  int*    sFps = (int*)  (smem + N_ * 16);             // [1024]
  float*  redv = (float*)(smem + N_ * 16 + 4096);      // [32]
  int*    redi = (int*)  (smem + N_ * 16 + 4096 + 128);// [32]
  int*    sWin = (int*)  (smem + N_ * 16 + 4096 + 256);

  const int b = blockIdx.x, tid = threadIdx.x;
  const float* bx = xyz + (size_t)b * N_ * 3;

  // Async-stage the packed [N][3] cloud into padded [N] float4 slots.
  {
    const unsigned base = lds_addr_of(pts);
    for (int i = tid; i < N_ * 3; i += 1024) {
      int p = i / 3, c = i - p * 3;
      async_copy_dword(base + p * 16 + c * 4, bx + i);
    }
    wait_async0();
  }
  if (tid == 0) sFps[0] = 0;
  __syncthreads();

  float mind[16];
#pragma unroll
  for (int j = 0; j < 16; ++j) mind[j] = 1e10f;

  float4 lp = pts[0];                         // last selected point (broadcast)
  for (int it = 1; it < M_; ++it) {
    float bestv = -1.f; int besti = 0;
#pragma unroll
    for (int j = 0; j < 16; ++j) {
      int p = tid + (j << 10);
      float4 s = pts[p];
      float dx = s.x - lp.x, dy = s.y - lp.y, dz = s.z - lp.z;
      float d = dx * dx + dy * dy + dz * dz;
      float m = fminf(mind[j], d);
      mind[j] = m;
      if (m > bestv) { bestv = m; besti = p; }
    }
    // wave32 argmax (tie -> lowest index, matching jnp.argmax)
#pragma unroll
    for (int off = 16; off >= 1; off >>= 1) {
      float ov = __shfl_xor(bestv, off, 32);
      int   oi = __shfl_xor(besti, off, 32);
      if (ov > bestv || (ov == bestv && oi < besti)) { bestv = ov; besti = oi; }
    }
    if ((tid & 31) == 0) { redv[tid >> 5] = bestv; redi[tid >> 5] = besti; }
    __syncthreads();
    if (tid < 32) {
      float v = redv[tid]; int ii = redi[tid];
#pragma unroll
      for (int off = 16; off >= 1; off >>= 1) {
        float ov = __shfl_xor(v, off, 32);
        int   oi = __shfl_xor(ii, off, 32);
        if (ov > v || (ov == v && oi < ii)) { v = ov; ii = oi; }
      }
      if (tid == 0) { sWin[0] = ii; sFps[it] = ii; }
    }
    __syncthreads();
    lp = pts[sWin[0]];                        // LDS broadcast of new centroid
  }
  __syncthreads();
  {
    int ii = sFps[tid];
    float4 s = pts[ii];
    float* o = qxyz + (size_t)(b * M_ + tid) * 3;
    o[0] = s.x; o[1] = s.y; o[2] = s.z;
  }
}

// =====================================================================
// Kernel 2: exact kNN (top-32 smallest squared distances, ascending).
// One thread per query; sorted top-32 kept in LDS; support xyz tiles
// async-DMA'd into LDS and shared by all 128 queries in the block.
// =====================================================================
__global__ __launch_bounds__(128) void knn_kernel(const float* __restrict__ xyz,
                                                  const float* __restrict__ qxyz,
                                                  int* __restrict__ knn) {
  extern __shared__ __align__(16) char smem[];
  float4* tile = (float4*)smem;                  // [1024] = 16 KB
  float*  kd   = (float*)(smem + 16384);         // [32][128]
  int*    ki   = (int*)  (smem + 32768);         // [32][128]

  const int t  = threadIdx.x;
  const int gq = blockIdx.x * 128 + t;           // 1024 % 128 == 0 -> one batch/block
  const int b  = gq >> 10;
  const float* bx = xyz + (size_t)b * N_ * 3;
  const unsigned tbase = lds_addr_of(tile);

  const float qx = qxyz[(size_t)gq * 3 + 0];
  const float qy = qxyz[(size_t)gq * 3 + 1];
  const float qz = qxyz[(size_t)gq * 3 + 2];

#pragma unroll
  for (int j = 0; j < NS; ++j) { kd[j * 128 + t] = 1e30f; ki[j * 128 + t] = 0; }
  float worst = 1e30f;

  for (int tb = 0; tb < N_; tb += 1024) {
    __syncthreads();                             // readers of prev tile done
    for (int i = t; i < 1024 * 3; i += 128) {    // async-stage packed xyz tile
      int p = i / 3, c = i - p * 3;
      async_copy_dword(tbase + p * 16 + c * 4, bx + (size_t)tb * 3 + i);
    }
    wait_async0();
    __syncthreads();
    for (int p = 0; p < 1024; ++p) {
      float4 s = tile[p];
      float dx = s.x - qx, dy = s.y - qy, dz = s.z - qz;
      float d = dx * dx + dy * dy + dz * dz;
      if (d < worst) {                       // rare after warm-up (~230x total)
        int j = NS - 1;
        while (j > 0 && kd[(j - 1) * 128 + t] > d) {
          kd[j * 128 + t] = kd[(j - 1) * 128 + t];
          ki[j * 128 + t] = ki[(j - 1) * 128 + t];
          --j;
        }
        kd[j * 128 + t] = d; ki[j * 128 + t] = tb + p;
        worst = kd[(NS - 1) * 128 + t];
      }
    }
  }
#pragma unroll
  for (int j = 0; j < NS; ++j) knn[(size_t)gq * NS + j] = ki[j * 128 + t];
}

// =====================================================================
// Kernel 3: fused gather + MLP(67->128->256) + neighborhood maxpool,
// via v_wmma_f32_16x16x32_f16 (f16 operands, f32 accumulate).
// Weight (B) fragments live in registers for the whole block; A tiles
// are shared across all accumulators so each ds-load pair feeds 2-4
// independent WMMAs (hides the WMMA RAW hazard + LDS latency).
// =====================================================================

// A operand (16x32 f16 tile) from row-major LDS. ISA layout: lanes 0-15 hold
// row M=lane with K in {k0..k0+7, k0+16..k0+23}; lanes 16-31 the same row
// shifted by 8 in K. Two 16-byte ds loads per lane.
__device__ __forceinline__ v16h load_a_frag(const _Float16* base, int ld,
                                            int row0, int k0, int lane) {
  int half = lane >> 4;
  int m    = lane & 15;
  const _Float16* p = base + (row0 + m) * ld + k0 + half * 8;
  v8h lo = *(const v8h*)p;          // K = k0 + half*8 .. +7
  v8h hi = *(const v8h*)(p + 16);   // K = k0 + 16 + half*8 .. +7
  return __builtin_shufflevector(lo, hi, 0, 1, 2, 3, 4, 5, 6, 7,
                                 8, 9, 10, 11, 12, 13, 14, 15);
}

// B operand (32x16 f16 tile) from W^T stored [N][K] row-major in LDS.
// ISA layout: lane n<16 holds column N=n, K in {k0..k0+7, k0+16..k0+23};
// lanes 16-31 the same shifted by 8 in K (matches the A fragment).
__device__ __forceinline__ v16h load_b_frag(const _Float16* baseT, int ldk,
                                            int n0, int k0, int lane) {
  int kh = lane >> 4;
  int n  = n0 + (lane & 15);
  const _Float16* p = baseT + n * ldk + k0 + kh * 8;
  v8h lo = *(const v8h*)p;
  v8h hi = *(const v8h*)(p + 16);
  return __builtin_shufflevector(lo, hi, 0, 1, 2, 3, 4, 5, 6, 7,
                                 8, 9, 10, 11, 12, 13, 14, 15);
}

__device__ __forceinline__ v8f wmma_f16(v16h a, v16h b, v8f c) {
  return __builtin_amdgcn_wmma_f32_16x16x32_f16(false, a, false, b,
                                                (short)0, c, false, false);
}

__global__ __launch_bounds__(256) void mlp_kernel(const float* __restrict__ xyz,
                                                  const float* __restrict__ feat,
                                                  const float* __restrict__ qxyz,
                                                  const int* __restrict__ knn,
                                                  const float* __restrict__ W1,
                                                  const float* __restrict__ b1,
                                                  const float* __restrict__ W2,
                                                  const float* __restrict__ b2,
                                                  float* __restrict__ outf) {
  extern __shared__ __align__(16) char smem[];
  _Float16* sW1T = (_Float16*)(smem);            // [128][96]  f16 (K padded 67->96)
  _Float16* sW2T = (_Float16*)(smem + 24576);    // [256][128] f16
  _Float16* sA1  = (_Float16*)(smem + 90112);    // [32][96]   grouped inputs
  _Float16* sA2  = (_Float16*)(smem + 96256);    // [32][128]  relu(h1)
  float*    sB1  = (float*)   (smem + 104448);   // [128]
  float*    sB2  = (float*)   (smem + 104960);   // [256]
  float*    sQ   = (float*)   (smem + 105984);   // [4]
  int*      sIdx = (int*)     (smem + 106000);   // [32]

  const int tid  = threadIdx.x;
  const int lane = tid & 31;
  const int w    = tid >> 5;                     // 8 waves

  // Stage weights once per block (f32 -> f16, transposed to B layout).
  for (int e = tid; e < 128 * 96; e += 256) {
    int n = e / 96, k = e - n * 96;
    sW1T[e] = (k < 67) ? (_Float16)W1[k * 128 + n] : (_Float16)0.f;
  }
  for (int e = tid; e < 256 * 128; e += 256) {
    int n = e >> 7, k = e & 127;
    sW2T[e] = (_Float16)W2[k * 256 + n];
  }
  if (tid < 128) sB1[tid] = b1[tid];
  sB2[tid] = b2[tid];
  __syncthreads();

  // Hoist this wave's weight fragments + biases into registers: reused
  // for all 8 queries of the block. 11 x v16h = 88 VGPRs.
  const int n1 = w * 16;                         // GEMM1 column tile
  const int n2 = w * 32;                         // GEMM2 column pair
  v16h b1f[3];
#pragma unroll
  for (int kk = 0; kk < 3; ++kk) b1f[kk] = load_b_frag(sW1T, 96, n1, kk * 32, lane);
  v16h b2f[2][4];
#pragma unroll
  for (int ct = 0; ct < 2; ++ct)
#pragma unroll
    for (int kk = 0; kk < 4; ++kk)
      b2f[ct][kk] = load_b_frag(sW2T, 128, n2 + ct * 16, kk * 32, lane);
  const float bias1   = sB1[n1 + (lane & 15)];
  const float bias2_0 = sB2[n2 + (lane & 15)];
  const float bias2_1 = sB2[n2 + 16 + (lane & 15)];

#pragma unroll 1
  for (int q = 0; q < 8; ++q) {
    const int gq = blockIdx.x * 8 + q;
    const int b  = gq >> 10;
    const int mq = gq & 1023;

    __syncthreads();                             // prior GEMM2 reads done
    if (tid < 32) sIdx[tid] = knn[(size_t)gq * 32 + tid];
    if (tid < 3)  sQ[tid]   = qxyz[(size_t)gq * 3 + tid];
    __syncthreads();

    // Build A1 = [rel_xyz | neighbor features | 0-pad] as f16, 32 x 96.
    for (int e = tid; e < 32 * 96; e += 256) {
      int s = e / 96, c = e - s * 96;
      int ni = sIdx[s];
      float v;
      if (c < 3)       v = xyz[((size_t)b * N_ + ni) * 3 + c] - sQ[c];
      else if (c < 67) v = feat[((size_t)b * N_ + ni) * 64 + (c - 3)];
      else             v = 0.f;
      sA1[e] = (_Float16)v;
    }
    __syncthreads();

    // GEMM1: h1 = relu(A1 @ W1 + b1); this wave owns cols [n1, n1+16).
    // Two independent accumulators (row tiles) share each B fragment.
    {
      v8f acc[2] = {};
#pragma unroll
      for (int kk = 0; kk < 3; ++kk) {
        v16h a0 = load_a_frag(sA1, 96, 0,  kk * 32, lane);
        v16h a1 = load_a_frag(sA1, 96, 16, kk * 32, lane);
        acc[0] = wmma_f16(a0, b1f[kk], acc[0]);
        acc[1] = wmma_f16(a1, b1f[kk], acc[1]);
      }
#pragma unroll
      for (int rt = 0; rt < 2; ++rt) {
#pragma unroll
        for (int r = 0; r < 8; ++r) {
          float v = acc[rt][r] + bias1;
          v = v > 0.f ? v : 0.f;
          int row = rt * 16 + ((lane < 16) ? r : r + 8);   // C-tile layout
          sA2[row * 128 + n1 + (lane & 15)] = (_Float16)v;
        }
      }
    }
    __syncthreads();

    // GEMM2 + maxpool; this wave owns cols [n2, n2+32). Each A-frag pair
    // feeds 4 independent accumulators (2 col tiles x 2 row tiles).
    {
      v8f acc[2][2] = {};
#pragma unroll
      for (int kk = 0; kk < 4; ++kk) {
        v16h a0 = load_a_frag(sA2, 128, 0,  kk * 32, lane);
        v16h a1 = load_a_frag(sA2, 128, 16, kk * 32, lane);
#pragma unroll
        for (int ct = 0; ct < 2; ++ct) {
          acc[ct][0] = wmma_f16(a0, b2f[ct][kk], acc[ct][0]);
          acc[ct][1] = wmma_f16(a1, b2f[ct][kk], acc[ct][1]);
        }
      }
      // max over rows: 8 accumulator regs cover M={0..7}/{8..15} per lane
      // half; bias is per-column so relu(max+b) == max(relu(x+b)).
#pragma unroll
      for (int ct = 0; ct < 2; ++ct) {
        float m = -1e30f;
#pragma unroll
        for (int r = 0; r < 8; ++r) {
          m = fmaxf(m, acc[ct][0][r]);
          m = fmaxf(m, acc[ct][1][r]);
        }
        m += (ct == 0) ? bias2_0 : bias2_1;
        m = fmaxf(m, 0.f);
        m = fmaxf(m, __shfl_xor(m, 16, 32));     // combine M-halves
        if (lane < 16)
          outf[(size_t)(b * 1024 + mq) * 256 + n2 + ct * 16 + lane] = m;
      }
    }
  }
}

// =====================================================================
extern "C" void kernel_launch(void* const* d_in, const int* in_sizes, int n_in,
                              void* d_out, int out_size, void* d_ws, size_t ws_size,
                              hipStream_t stream) {
  (void)in_sizes; (void)n_in; (void)out_size; (void)ws_size;
  const float* sxyz  = (const float*)d_in[0];
  const float* sfeat = (const float*)d_in[1];
  const float* W1    = (const float*)d_in[2];
  const float* b1    = (const float*)d_in[3];
  const float* W2    = (const float*)d_in[4];
  const float* b2    = (const float*)d_in[5];

  float* qxyz = (float*)d_out;                     // [B,1024,3]
  float* feat = qxyz + (size_t)B_ * M_ * 3;        // [B,1024,256]
  int*   knn  = (int*)d_ws;                        // [B*1024, 32] = 512 KB

  fps_kernel<<<B_, 1024, 266512, stream>>>(sxyz, qxyz);
  knn_kernel<<<(B_ * M_) / 128, 128, 49152, stream>>>(sxyz, qxyz, knn);
  mlp_kernel<<<(B_ * M_) / 8, 256, 106128, stream>>>(sxyz, sfeat, qxyz, knn,
                                                     W1, b1, W2, b2, feat);
}
